// AttentionHead_74629351735817
// MI455X (gfx1250) — compile-verified, matrix operand order checked by reading, not run on hardware
//
#include <hip/hip_runtime.h>

// ---------------- problem constants ----------------
#define BB      8
#define TT      2048
#define DM      1024
#define DI      128
#define NQ      384            // 3 * DI
#define BT      (BB * TT)      // 16384 rows
#define SCALE   32.0f          // sqrt(D_MODEL)
#define LN10000 9.210340371976184f
#define PSTR    40             // padded LDS stride (halves): 80B, b128-aligned

typedef __attribute__((ext_vector_type(16))) _Float16 v16h;
typedef __attribute__((ext_vector_type(8)))  _Float16 v8h;
typedef __attribute__((ext_vector_type(8)))  float    v8f;
typedef __attribute__((ext_vector_type(4)))  unsigned int v4u;
typedef __attribute__((ext_vector_type(8)))  int      v8i;
typedef __attribute__((ext_vector_type(4)))  int      v4i;

#if defined(__has_builtin)
#  if __has_builtin(__builtin_amdgcn_tensor_load_to_lds) && __has_builtin(__builtin_amdgcn_s_wait_tensorcnt)
#    define USE_TDM 1
#  endif
#endif
#ifndef USE_TDM
#  define USE_TDM 0
#endif

union FragH {
    v16h v;
    v8h  h8[2];
    _Float16 h[16];
};

__device__ __forceinline__ v8f wmma_f16(v16h a, v16h b, v8f c) {
    return __builtin_amdgcn_wmma_f32_16x16x32_f16(
        false, a, false, b, (short)0, c, false, false);
}

#if USE_TDM
// Issue one TDM load of a [rows x 128] f16 tile (row-major) into LDS.
// D# per CDNA5 ISA.
__device__ __forceinline__ void tdm_load_tile(unsigned int lds_off,
                                              const _Float16* gptr,
                                              int tensor_rows) {
    unsigned long long ga = (unsigned long long)(size_t)gptr;
    v4u g0;
    g0[0] = 1u;                                            // count=1, user mode
    g0[1] = lds_off;                                       // lds_addr
    g0[2] = (unsigned int)ga;                              // global_addr[31:0]
    g0[3] = (unsigned int)((ga >> 32) & 0x01FFFFFFu) | (2u << 30); // addr[56:32] | type=2
    v8i g1;
    g1[0] = (int)(1u << 16);                               // data_size = 1 (2 bytes)
    g1[1] = (int)(((unsigned)DI & 0xFFFFu) << 16);         // tensor_dim0[15:0]=128
    g1[2] = (int)(((unsigned)tensor_rows & 0xFFFFu) << 16);// dim0 hi=0 | tensor_dim1 lo
    g1[3] = (int)((unsigned)DI << 16);                     // dim1 hi=0 | tile_dim0=128
    g1[4] = 32;                                            // tile_dim1=32, tile_dim2=0
    g1[5] = DI;                                            // tensor_dim0_stride=128
    g1[6] = 0;
    g1[7] = 0;
    v4i z = {};
#if __clang_major__ >= 23
    v8i z8 = {};
    __builtin_amdgcn_tensor_load_to_lds(g0, g1, z, z, z8, 0);
#else
    __builtin_amdgcn_tensor_load_to_lds(g0, g1, z, z, 0);
#endif
}
#endif

// ============================================================
// Kernel 0: f32 -> f16 conversion (x and W_qkv), 8 elems/thread
// ============================================================
__global__ __launch_bounds__(256) void cvt_f16_kernel(
    const float* __restrict__ src, _Float16* __restrict__ dst)
{
    const size_t i = ((size_t)blockIdx.x * 256 + threadIdx.x) * 8;
    float4 u = *(const float4*)(src + i);
    float4 w = *(const float4*)(src + i + 4);
    v8h o;
    o[0] = (_Float16)u.x; o[1] = (_Float16)u.y; o[2] = (_Float16)u.z; o[3] = (_Float16)u.w;
    o[4] = (_Float16)w.x; o[5] = (_Float16)w.y; o[6] = (_Float16)w.z; o[7] = (_Float16)w.w;
    *(v8h*)(dst + i) = o;
}

// ============================================================
// Kernel 1: QKV projection, f16 WMMA. One wave per 16x48 strip;
// fragments double-buffered in registers so each WMMA waits on
// loads issued a full k-step earlier (latency hidden in-wave).
// ============================================================
__global__ __launch_bounds__(128) void qkv_gemm_kernel(
    const _Float16* __restrict__ xh, const _Float16* __restrict__ wh,
    float* __restrict__ qkv)
{
    const int lane = threadIdx.x & 31;
    const int wave = threadIdx.x >> 5;
    const int tile = blockIdx.x * 4 + wave;   // 0 .. 8191
    const int tm    = tile >> 3;              // 1024 M tiles
    const int strip = tile & 7;               // 8 strips of 3 N-tiles

    const int half = (lane >> 4) & 1;
    const int nl   = lane & 15;
    const int koff = half * 8;

    const _Float16* arow = xh + (size_t)(tm * 16 + nl) * DM;
    const _Float16* brow[3];
    #pragma unroll
    for (int i = 0; i < 3; ++i)
        brow[i] = wh + (size_t)((strip * 3 + i) * 16 + nl) * DM;

    v8f c[3];
    #pragma unroll
    for (int i = 0; i < 3; ++i) { v8f z = {}; c[i] = z; }

    FragH a[2], bf[2][3];
    // prologue: fill buffer 0 with k=0 fragments
    a[0].h8[0] = *(const v8h*)(arow + koff);
    a[0].h8[1] = *(const v8h*)(arow + 16 + koff);
    #pragma unroll
    for (int i = 0; i < 3; ++i) {
        bf[0][i].h8[0] = *(const v8h*)(brow[i] + koff);
        bf[0][i].h8[1] = *(const v8h*)(brow[i] + 16 + koff);
    }

    for (int k = 0; k < DM; k += 32) {
        const int cur = (k >> 5) & 1;
        const int nxt = cur ^ 1;
        if (k + 32 < DM) {
            const int kn = k + 32;
            a[nxt].h8[0] = *(const v8h*)(arow + kn + koff);
            a[nxt].h8[1] = *(const v8h*)(arow + kn + 16 + koff);
            #pragma unroll
            for (int i = 0; i < 3; ++i) {
                bf[nxt][i].h8[0] = *(const v8h*)(brow[i] + kn + koff);
                bf[nxt][i].h8[1] = *(const v8h*)(brow[i] + kn + 16 + koff);
            }
        }
        #pragma unroll
        for (int i = 0; i < 3; ++i)
            c[i] = wmma_f16(a[cur].v, bf[cur][i].v, c[i]);
    }

    #pragma unroll
    for (int i = 0; i < 3; ++i) {
        float* orow = qkv + (size_t)(tm * 16) * NQ + (strip * 3 + i) * 16 + nl;
        #pragma unroll
        for (int r = 0; r < 8; ++r)
            orow[(size_t)(r + half * 8) * NQ] = c[i][r];
    }
}

// ============================================================
// Kernel 2: RoPE + L2-normalize + sqk scale.
// Emits Qh,Kh row-major f16 and V transposed: Vt[b][d][t] f16.
// ============================================================
__device__ __forceinline__ float wave_sum(float v) {
    #pragma unroll
    for (int m = 16; m >= 1; m >>= 1) v += __shfl_xor(v, m, 32);
    return v;
}

__global__ __launch_bounds__(128) void rope_norm_kernel(
    const float* __restrict__ qkv, const float* __restrict__ sqk,
    _Float16* __restrict__ Qh, _Float16* __restrict__ Kh,
    _Float16* __restrict__ Vt)
{
    const int lane = threadIdx.x & 31;
    const int row  = blockIdx.x * 4 + (threadIdx.x >> 5);
    const int tpos = row & (TT - 1);
    const int bidx = row >> 11;               // row / TT
    const float* rq = qkv + (size_t)row * NQ;

    float q[4], k[4], vv[4], cs[4], sn[4];
    #pragma unroll
    for (int j = 0; j < 4; ++j) {
        const int d = lane + 32 * j;
        q[j]  = rq[d];
        k[j]  = rq[DI + d];
        vv[j] = rq[2 * DI + d];
        const int idx = d & 63;
        const float invf = __expf(-(float)idx * (2.0f / (float)DI) * LN10000);
        __sincosf((float)tpos * invf, &sn[j], &cs[j]);
    }
    float qr[4], kr[4];
    qr[0] = q[0] * cs[0] - q[2] * sn[0];
    qr[1] = q[1] * cs[1] - q[3] * sn[1];
    qr[2] = q[2] * cs[2] + q[0] * sn[2];
    qr[3] = q[3] * cs[3] + q[1] * sn[3];
    kr[0] = k[0] * cs[0] - k[2] * sn[0];
    kr[1] = k[1] * cs[1] - k[3] * sn[1];
    kr[2] = k[2] * cs[2] + k[0] * sn[2];
    kr[3] = k[3] * cs[3] + k[1] * sn[3];

    float ssq = qr[0]*qr[0] + qr[1]*qr[1] + qr[2]*qr[2] + qr[3]*qr[3];
    float ssk = kr[0]*kr[0] + kr[1]*kr[1] + kr[2]*kr[2] + kr[3]*kr[3];
    const float rq_n = rsqrtf(wave_sum(ssq));
    const float rk_n = rsqrtf(wave_sum(ssk));

    #pragma unroll
    for (int j = 0; j < 4; ++j) {
        const int d = lane + 32 * j;
        const float sc = sqk[d];
        Qh[(size_t)row * DI + d] = (_Float16)(qr[j] * rq_n * sc);
        Kh[(size_t)row * DI + d] = (_Float16)(kr[j] * rk_n * sc);
        Vt[((size_t)(bidx * DI + d)) * TT + tpos] = (_Float16)vv[j];
    }
}

// ============================================================
// Kernel 3: causal flash attention, one wave per 16-query tile.
// No online max (nGPT bounds |score| <= 1/32). K blocks double-
// buffered into LDS via TDM when available.
// ============================================================
__global__ __launch_bounds__(32) void flash_attn_kernel(
    const _Float16* __restrict__ Qh, const _Float16* __restrict__ Kh,
    const _Float16* __restrict__ Vt, float* __restrict__ out)
{
    __shared__ _Float16 Pl[16 * PSTR];
#if USE_TDM
    __shared__ _Float16 Klds[2][32 * DI];
#endif
    const int lane = threadIdx.x;
    const int half = lane >> 4;
    const int nl   = lane & 15;
    const int koff = half * 8;

    const int b  = blockIdx.x >> 7;           // 128 query tiles per batch
    const int qt = blockIdx.x & 127;
    const int qrow0 = b * TT + qt * 16;

    // Hoisted Q A-fragments: 16x128 as 4 chunks of 16x32
    FragH qf[4];
    const _Float16* qrow = Qh + (size_t)(qrow0 + nl) * DI;
    #pragma unroll
    for (int cch = 0; cch < 4; ++cch) {
        qf[cch].h8[0] = *(const v8h*)(qrow + cch * 32 + koff);
        qf[cch].h8[1] = *(const v8h*)(qrow + cch * 32 + 16 + koff);
    }

    v8f acc[8];
    float psum[8];
    #pragma unroll
    for (int dt = 0; dt < 8; ++dt) { v8f z = {}; acc[dt] = z; }
    #pragma unroll
    for (int r = 0; r < 8; ++r) psum[r] = 0.0f;

    const int kbmax = (qt * 16 + 15) >> 5;    // inclusive last 32-key block
    const _Float16* kbase = Kh + (size_t)b * TT * DI;
    const _Float16* vbase = Vt + (size_t)b * DI * TT;

#if USE_TDM
    tdm_load_tile((unsigned int)(size_t)&Klds[0][0], kbase, TT);
#endif

    for (int kb = 0; kb <= kbmax; ++kb) {
        const int key0 = kb * 32;

#if USE_TDM
        if (kb < kbmax) {
            tdm_load_tile((unsigned int)(size_t)&Klds[(kb + 1) & 1][0],
                          kbase + (size_t)(key0 + 32) * DI, TT - (key0 + 32));
            __builtin_amdgcn_s_wait_tensorcnt(1);   // current block ready
        } else {
            __builtin_amdgcn_s_wait_tensorcnt(0);
        }
        const _Float16* kblk = &Klds[kb & 1][0];
#else
        const _Float16* kblk = kbase + (size_t)key0 * DI;
        if (kb < kbmax)
            __builtin_prefetch(kbase + (size_t)(key0 + 32) * DI, 0, 0);
#endif

        // ---- S = Q * K^T : two 16x16 tiles over 32 keys.
        // Pre-load both nt-tiles' first-chunk fragments so the 8
        // WMMAs have independent load streams to overlap with.
        v8f s[2];
        #pragma unroll
        for (int nt = 0; nt < 2; ++nt) {
            const _Float16* krow = kblk + (size_t)(nt * 16 + nl) * DI;
            FragH bf[2];
            bf[0].h8[0] = *(const v8h*)(krow + koff);
            bf[0].h8[1] = *(const v8h*)(krow + 16 + koff);
            v8f cc = {};
            #pragma unroll
            for (int cch = 0; cch < 4; ++cch) {
                if (cch + 1 < 4) {
                    const int kn = (cch + 1) * 32;
                    bf[(cch + 1) & 1].h8[0] = *(const v8h*)(krow + kn + koff);
                    bf[(cch + 1) & 1].h8[1] = *(const v8h*)(krow + kn + 16 + koff);
                }
                cc = wmma_f16(qf[cch].v, bf[cch & 1].v, cc);
            }
            s[nt] = cc;
        }

        // ---- softmax numerator: p = exp(32*s); mask only on diagonal ----
        const bool needmask = (key0 + 31 > qt * 16);
        #pragma unroll
        for (int r = 0; r < 8; ++r) {
            float p0 = __expf(s[0][r] * SCALE);
            float p1 = __expf(s[1][r] * SCALE);
            if (needmask) {
                const int qg = qt * 16 + r + half * 8;
                if (key0 + nl      > qg) p0 = 0.0f;
                if (key0 + 16 + nl > qg) p1 = 0.0f;
            }
            psum[r] += p0 + p1;
            s[0][r] = p0; s[1][r] = p1;
        }

        __syncthreads();   // also orders prior iteration's Pl reads (WAR)

        // ---- relayout P (C layout -> f16 A layout) through LDS ----
        #pragma unroll
        for (int r = 0; r < 8; ++r) {
            const int m = r + half * 8;
            Pl[m * PSTR + nl]      = (_Float16)s[0][r];
            Pl[m * PSTR + 16 + nl] = (_Float16)s[1][r];
        }
        __syncthreads();

        FragH pf;
        pf.h8[0] = *(const v8h*)(&Pl[nl * PSTR + koff]);
        pf.h8[1] = *(const v8h*)(&Pl[nl * PSTR + 16 + koff]);

        // ---- acc += P(16x32) * V(32x16) per d-tile; V from Vt (L2) ----
        #pragma unroll
        for (int dt = 0; dt < 8; ++dt) {
            const _Float16* vrow = vbase + (size_t)(dt * 16 + nl) * TT + key0;
            FragH vf;
            vf.h8[0] = *(const v8h*)(vrow + koff);
            vf.h8[1] = *(const v8h*)(vrow + 16 + koff);
            acc[dt] = wmma_f16(pf.v, vf.v, acc[dt]);
        }
    }

    // ---- epilogue: single row-sum reduction, then out = acc / l ----
    float il[8];
    #pragma unroll
    for (int r = 0; r < 8; ++r) {
        float ps = psum[r];
        #pragma unroll
        for (int msk = 8; msk >= 1; msk >>= 1)   // rows live in one 16-lane half
            ps += __shfl_xor(ps, msk, 32);
        il[r] = 1.0f / ps;
    }
    float* orow = out + (size_t)qrow0 * DI;
    #pragma unroll
    for (int dt = 0; dt < 8; ++dt) {
        #pragma unroll
        for (int r = 0; r < 8; ++r) {
            const int m = r + half * 8;
            orow[(size_t)m * DI + dt * 16 + nl] = acc[dt][r] * il[r];
        }
    }
}

// ============================================================
extern "C" void kernel_launch(void* const* d_in, const int* in_sizes, int n_in,
                              void* d_out, int out_size, void* d_ws, size_t ws_size,
                              hipStream_t stream) {
    const float* x   = (const float*)d_in[0];   // [B,T,1024]
    const float* Wq  = (const float*)d_in[1];   // [384,1024]
    const float* sqk = (const float*)d_in[2];   // [128]

    // workspace layout
    char* ws = (char*)d_ws;
    _Float16* xh  = (_Float16*)ws;                                   // 32 MB
    _Float16* wh  = (_Float16*)(ws + (size_t)BT * DM * 2);           // 768 KB
    float*    qkv = (float*)   (ws + (size_t)BT * DM * 2 + (size_t)NQ * DM * 2); // 24 MB
    _Float16* Qh  = (_Float16*)((char*)qkv + (size_t)BT * NQ * 4);   // 4 MB
    _Float16* Kh  = Qh + (size_t)BT * DI;                            // 4 MB
    _Float16* Vt  = Kh + (size_t)BT * DI;                            // 4 MB

    // 0) f32 -> f16 conversions
    cvt_f16_kernel<<<(BT * DM) / (256 * 8), 256, 0, stream>>>(x, xh);
    cvt_f16_kernel<<<(NQ * DM) / (256 * 8), 256, 0, stream>>>(Wq, wh);

    // 1) QKV projection: 1024 M-tiles x 8 strips, 4 waves/block
    qkv_gemm_kernel<<<(BT / 16) * 8 / 4, 128, 0, stream>>>(xh, wh, qkv);

    // 2) RoPE + normalize + scale (+ V transpose)
    rope_norm_kernel<<<BT / 4, 128, 0, stream>>>(qkv, sqk, Qh, Kh, Vt);

    // 3) causal flash attention: 1 wave per 16-query tile
    flash_attn_kernel<<<BB * (TT / 16), 32, 0, stream>>>(Qh, Kh, Vt, (float*)d_out);
}
